// ContextWindowAttention_58617713655994
// MI455X (gfx1250) — matrix operand order repaired and checked by reading, not compiled, not run on hardware
//
#include <hip/hip_runtime.h>

#define WSN 49
#define CDIM 128
#define NPAD 64
#define SCALEQ 0.17677669529663687f  // 1/sqrt(32)

typedef _Float16 h16v __attribute__((ext_vector_type(16)));
typedef _Float16 h8v  __attribute__((ext_vector_type(8)));
typedef _Float16 h4v  __attribute__((ext_vector_type(4)));
typedef float    f8v  __attribute__((ext_vector_type(8)));
typedef float    f4v  __attribute__((ext_vector_type(4)));

// A-operand (16xK tile, 16-bit): lane<16 rows 0..15 hold K {s..s+7, 16+s..16+s+7}, s=8*(lane>=16)
__device__ __forceinline__ h16v load_a(const _Float16* base, int ld, int lane) {
    int r = lane & 15;
    int s = (lane >> 4) << 3;
    const _Float16* p = base + r * ld + s;
    h8v lo = *(const h8v*)(p);
    h8v hi = *(const h8v*)(p + 16);
    h16v o;
#pragma unroll
    for (int i = 0; i < 8; ++i) { o[i] = lo[i]; o[i + 8] = hi[i]; }
    return o;
}

// B-operand (Kx16 tile, 16-bit), fed from an N-major array (row = n, contiguous K):
// lane<16: col n=lane, K {0..7, 16..23}; lane>=16: col n=lane-16, K {8..15, 24..31}
__device__ __forceinline__ h16v load_b(const _Float16* base, int ld, int lane) {
    int r = lane & 15;
    int s = (lane >> 4) << 3;
    const _Float16* p = base + r * ld + s;
    h8v lo = *(const h8v*)(p);
    h8v hi = *(const h8v*)(p + 16);
    h16v o;
#pragma unroll
    for (int i = 0; i < 8; ++i) { o[i] = lo[i]; o[i + 8] = hi[i]; }
    return o;
}

__device__ __forceinline__ f8v wmma_f16(h16v a, h16v b, f8v c) {
    return __builtin_amdgcn_wmma_f32_16x16x32_f16(false, a, false, b, (short)0, c, false, false);
}

// -------- prep kernels --------

__global__ void transpose_f32_to_f16(const float* __restrict__ src, _Float16* __restrict__ dst,
                                     int K, int Nn) {
    int id = blockIdx.x * blockDim.x + threadIdx.x;
    if (id >= K * Nn) return;
    int k = id / Nn, n = id % Nn;
    dst[n * K + k] = (_Float16)src[id];
}

// biasP[h][n][m] (64x64 padded): pad columns get -1e30 so softmax masks them for free
__global__ void bias_prep(const float* __restrict__ table, const int* __restrict__ rel,
                          float* __restrict__ biasP) {
    int id = blockIdx.x * blockDim.x + threadIdx.x;
    if (id >= 4 * 64 * 64) return;
    int h = id >> 12;
    int n = (id >> 6) & 63;
    int m = id & 63;
    float v;
    if (m >= WSN)      v = -1e30f;
    else if (n >= WSN) v = 0.0f;
    else               v = table[rel[n * WSN + m] * 4 + h];
    biasP[id] = v;
}

// -------- fused attention kernel: one block per window --------

__global__ __launch_bounds__(256)
void attn_fused(const float* __restrict__ x, const float* __restrict__ y,
                const float* __restrict__ mask,
                const float* __restrict__ bq, const float* __restrict__ bkv,
                const float* __restrict__ bp,
                const _Float16* __restrict__ WqT, const _Float16* __restrict__ WkvT,
                const _Float16* __restrict__ WpT, const float* __restrict__ biasP,
                float* __restrict__ out, int nW) {
    __shared__ __align__(16) _Float16 lds[32768];       // 64 KB
    _Float16* xh = lds;                                 // [64][128]; later vT [128][64]
    _Float16* yh = lds + 8192;                          // [64][128]; later sc+ph
    _Float16* qh = lds + 16384;                         // [64][128]
    _Float16* kh = lds + 24576;                         // [64][128]; later oh
    _Float16* vT = xh;
    _Float16* sc = yh;                                  // [64][64] f16 scores
    _Float16* ph = yh + 4096;                           // [64][64] f16 probs
    _Float16* oh = kh;                                  // [64][128] pre-proj output

    const int tid  = threadIdx.x;
    const int lane = tid & 31;
    const int w    = tid >> 5;          // wave id 0..7
    const int b    = blockIdx.x;
    const int ln   = lane & 15;
    const int mb   = (lane >> 4) << 3;  // row offset 0/8 for C/D tiles

    // ---- stage A: x,y -> f16 LDS (rows >= 49 zero-padded) ----
    {
        const f4v* xg = (const f4v*)x;
        const f4v* yg = (const f4v*)y;
#pragma unroll
        for (int it = 0; it < 8; ++it) {
            int f   = tid + it * 256;   // 0..2047
            int row = f >> 5;
            int c4  = f & 31;
            f4v vx = {0.f, 0.f, 0.f, 0.f}, vy = {0.f, 0.f, 0.f, 0.f};
            if (row < WSN) {
                size_t g = ((size_t)b * WSN + row) * 32 + c4;
                vx = xg[g];
                vy = yg[g];
            }
            h4v hx = {(_Float16)vx[0], (_Float16)vx[1], (_Float16)vx[2], (_Float16)vx[3]};
            h4v hy = {(_Float16)vy[0], (_Float16)vy[1], (_Float16)vy[2], (_Float16)vy[3]};
            *(h4v*)(xh + row * 128 + c4 * 4) = hx;
            *(h4v*)(yh + row * 128 + c4 * 4) = hy;
        }
    }
    __syncthreads();

    // ---- stage B: q = x@Wq, k = y@Wkv[:, :128]  (wave w owns output-col tile c = w) ----
    {
        int c = w;
        h16v bw[4];
#pragma unroll
        for (int kk = 0; kk < 4; ++kk) bw[kk] = load_b(WqT + (16 * c) * 128 + 32 * kk, 128, lane);
        float bias = bq[16 * c + ln];
#pragma unroll
        for (int r = 0; r < 4; ++r) {
            f8v acc = {};
#pragma unroll
            for (int kk = 0; kk < 4; ++kk) {
                h16v a = load_a(xh + (16 * r) * 128 + 32 * kk, 128, lane);
                acc = wmma_f16(a, bw[kk], acc);
            }
#pragma unroll
            for (int i = 0; i < 8; ++i)
                qh[(16 * r + mb + i) * 128 + 16 * c + ln] = (_Float16)((acc[i] + bias) * SCALEQ);
        }
#pragma unroll
        for (int kk = 0; kk < 4; ++kk) bw[kk] = load_b(WkvT + (16 * c) * 128 + 32 * kk, 128, lane);
        bias = bkv[16 * c + ln];
#pragma unroll
        for (int r = 0; r < 4; ++r) {
            f8v acc = {};
#pragma unroll
            for (int kk = 0; kk < 4; ++kk) {
                h16v a = load_a(yh + (16 * r) * 128 + 32 * kk, 128, lane);
                acc = wmma_f16(a, bw[kk], acc);
            }
#pragma unroll
            for (int i = 0; i < 8; ++i)
                kh[(16 * r + mb + i) * 128 + 16 * c + ln] = (_Float16)(acc[i] + bias);
        }
    }
    __syncthreads();

    // ---- stage B2: v = y@Wkv[:, 128:], stored transposed vT[d][m] (overwrites xh) ----
    {
        int c = w;
        h16v bw[4];
#pragma unroll
        for (int kk = 0; kk < 4; ++kk)
            bw[kk] = load_b(WkvT + (128 + 16 * c) * 128 + 32 * kk, 128, lane);
        float bias = bkv[128 + 16 * c + ln];
#pragma unroll
        for (int r = 0; r < 4; ++r) {
            f8v acc = {};
#pragma unroll
            for (int kk = 0; kk < 4; ++kk) {
                h16v a = load_a(yh + (16 * r) * 128 + 32 * kk, 128, lane);
                acc = wmma_f16(a, bw[kk], acc);
            }
            h8v st;
#pragma unroll
            for (int i = 0; i < 8; ++i) st[i] = (_Float16)(acc[i] + bias);
            *(h8v*)(vT + (16 * c + ln) * 64 + 16 * r + mb) = st;
        }
    }
    __syncthreads();

    // ---- stage C: per-head attention ----
    f8v oacc[4];
    const int w0 = b % nW;
    for (int h = 0; h < 4; ++h) {
        // scores S = q_h @ k_h^T : 16 tiles, wave w does t = w, w+8 (K=D=32, one WMMA each)
#pragma unroll
        for (int tt = 0; tt < 2; ++tt) {
            int t = w + 8 * tt;
            int r = t >> 2, c = t & 3;
            h16v a  = load_a(qh + (16 * r) * 128 + 32 * h, 128, lane);
            h16v bb = load_b(kh + (16 * c) * 128 + 32 * h, 128, lane);
            f8v acc = {};
            acc = wmma_f16(a, bb, acc);
#pragma unroll
            for (int i = 0; i < 8; ++i)
                sc[(16 * r + mb + i) * 64 + 16 * c + ln] = (_Float16)acc[i];
        }
        __syncthreads();

        // softmax: one thread per row; bias (with -1e30 pad cols) + window mask
        if (tid < 64) {
            int n = tid;
            const _Float16* srow = sc + n * 64;
            const float* brow = biasP + h * 4096 + n * 64;
            const float* mrow = mask + (size_t)w0 * (WSN * WSN) + n * WSN;
            bool mv = (n < WSN);
            float mx = -3.0e38f;
            for (int m = 0; m < 64; ++m) {
                float s = (float)srow[m] + brow[m];
                if (mv && m < WSN) s += mrow[m];
                mx = fmaxf(mx, s);
            }
            float sum = 0.f;
            for (int m = 0; m < 64; ++m) {
                float s = (float)srow[m] + brow[m];
                if (mv && m < WSN) s += mrow[m];
                float e = __expf(s - mx);
                sum += e;
                ph[n * 64 + m] = (_Float16)e;
            }
            float rs = 1.0f / sum;
            for (int m = 0; m < 64; ++m)
                ph[n * 64 + m] = (_Float16)((float)ph[n * 64 + m] * rs);
        }
        __syncthreads();

        // O_h = P @ v_h : 8 tiles, wave w owns (r = w>>1, c2 = w&1); keep in registers
        {
            int r = w >> 1, c2 = w & 1;
            f8v acc = {};
#pragma unroll
            for (int kk = 0; kk < 2; ++kk) {
                h16v a  = load_a(ph + (16 * r) * 64 + 32 * kk, 64, lane);
                h16v bb = load_b(vT + (32 * h + 16 * c2) * 64 + 32 * kk, 64, lane);
                acc = wmma_f16(a, bb, acc);
            }
            oacc[h] = acc;
        }
    }

    // ---- stage D: dump head outputs to LDS (kh region now dead), then out@Wp + bp ----
    {
        int r = w >> 1, c2 = w & 1;
#pragma unroll
        for (int h = 0; h < 4; ++h)
#pragma unroll
            for (int i = 0; i < 8; ++i)
                oh[(16 * r + mb + i) * 128 + 32 * h + 16 * c2 + ln] = (_Float16)oacc[h][i];
    }
    __syncthreads();
    {
        int c = w;
        h16v bw[4];
#pragma unroll
        for (int kk = 0; kk < 4; ++kk) bw[kk] = load_b(WpT + (16 * c) * 128 + 32 * kk, 128, lane);
        float bias = bp[16 * c + ln];
#pragma unroll
        for (int r = 0; r < 4; ++r) {
            f8v acc = {};
#pragma unroll
            for (int kk = 0; kk < 4; ++kk) {
                h16v a = load_a(oh + (16 * r) * 128 + 32 * kk, 128, lane);
                acc = wmma_f16(a, bw[kk], acc);
            }
#pragma unroll
            for (int i = 0; i < 8; ++i) {
                int n = 16 * r + mb + i;
                if (n < WSN)
                    out[((size_t)b * WSN + n) * 128 + 16 * c + ln] = acc[i] + bias;
            }
        }
    }
}

extern "C" void kernel_launch(void* const* d_in, const int* in_sizes, int n_in,
                              void* d_out, int out_size, void* d_ws, size_t ws_size,
                              hipStream_t stream) {
    const float* x     = (const float*)d_in[0];
    const float* y     = (const float*)d_in[1];
    const float* mask  = (const float*)d_in[2];
    const float* Wq    = (const float*)d_in[3];
    const float* bq    = (const float*)d_in[4];
    const float* Wkv   = (const float*)d_in[5];
    const float* bkv   = (const float*)d_in[6];
    const float* table = (const float*)d_in[7];
    const float* Wp    = (const float*)d_in[8];
    const float* bp    = (const float*)d_in[9];
    const int*   rel   = (const int*)d_in[10];
    float* out = (float*)d_out;

    char* ws = (char*)d_ws;
    _Float16* WqT   = (_Float16*)(ws);                       // 32 KB
    _Float16* WkvT  = (_Float16*)(ws + 32768);               // 64 KB
    _Float16* WpT   = (_Float16*)(ws + 32768 + 65536);       // 32 KB
    float*    biasP = (float*)(ws + 32768 + 65536 + 32768);  // 64 KB

    int Bn = in_sizes[0] / (WSN * CDIM);
    int nW = in_sizes[2] / (WSN * WSN);

    transpose_f32_to_f16<<<(128 * 128 + 255) / 256, 256, 0, stream>>>(Wq, WqT, 128, 128);
    transpose_f32_to_f16<<<(128 * 256 + 255) / 256, 256, 0, stream>>>(Wkv, WkvT, 128, 256);
    transpose_f32_to_f16<<<(128 * 128 + 255) / 256, 256, 0, stream>>>(Wp, WpT, 128, 128);
    bias_prep<<<(4 * 64 * 64 + 255) / 256, 256, 0, stream>>>(table, rel, biasP);

    attn_fused<<<Bn, 256, 0, stream>>>(x, y, mask, bq, bkv, bp,
                                       WqT, WkvT, WpT, biasP, out, nW);
}
